// IndicateVisitedPosition_60593398612562
// MI455X (gfx1250) — matrix operand order repaired and checked by reading, not compiled, not run on hardware
//
#include <hip/hip_runtime.h>
#include <cstdint>
#include <cstddef>

// ---------------------------------------------------------------------------
// IndicateVisitedPosition: sink-attention on MI455X (gfx1250), bf16 WMMA.
//
// ~200 GFLOP vs ~170 MB HBM -> compute bound: all five GEMMs run on
// v_wmma_f32_16x16x32_bf16 (f32 accum). Flash-attention 2-pass so the
// [S x S+1] score matrix never hits HBM; K/V (~10 MB/batch) stay L2-resident
// (192 MB), so the per-D-chunk score recompute is L2 traffic only.
//
// All WMMA B-operands (Wq/Wk/Wv/V_full) are stored in a fragment-linear
// swizzled layout so every fragment load is one aligned 32B load per lane
// (2x global_load_b128) instead of 16 strided d16 loads.
// ---------------------------------------------------------------------------

#define DI   1024   // model dim
#define PW   129    // W_POS+1 (real head dim)
#define PP   160    // head dim padded to multiple of 32 (5 K-chunks)
#define SQ   4096   // sequence length
#define KPAD 4128   // S+1 keys padded to multiple of 32
#define NB   4      // batch

typedef __attribute__((ext_vector_type(16))) __bf16 v16bf;
typedef __attribute__((ext_vector_type(8)))  __bf16 v8bf;
typedef __attribute__((ext_vector_type(8)))  float  v8f;

static __device__ __forceinline__ v8f wmma_bf16(v16bf a, v16bf b, v8f c) {
  // D = A(16x32 bf16) * B(32x16 bf16) + C(16x16 f32)
  return __builtin_amdgcn_wmma_f32_16x16x32_bf16(false, a, false, b,
                                                 (short)0, c, false, false);
}

// A fragment: 16x32 tile of a row-major bf16 matrix [*, lda].
// Contiguous K order (matches B loaders): lane L holds row M=L%16; lanes 0-15
// hold K=0..15, lanes 16-31 hold K=16..31. Two 16B vector loads.
static __device__ __forceinline__ v16bf load_a16(const __bf16* A, int lda,
                                                 int m0, int k0) {
  const int lane = threadIdx.x & 31;
  const __bf16* row =
      A + (size_t)(m0 + (lane & 15)) * lda + k0 + ((lane >> 4) * 16);
  v8bf lo = *(const v8bf*)(row);        // K offsets 0..7   (rel. to half)
  v8bf hi = *(const v8bf*)(row + 8);    // K offsets 8..15
  return __builtin_shufflevector(lo, hi, 0, 1, 2, 3, 4, 5, 6, 7, 8, 9, 10, 11,
                                 12, 13, 14, 15);
}

// B fragment from transposed view: B[k][n] = Kt[n0+n][k0+k] (row-major keys).
// 16 contiguous bf16 per lane, 32B-aligned -> single v16bf load.
static __device__ __forceinline__ v16bf load_bT16(const __bf16* Kt, int ldk,
                                                  int n0, int k0) {
  const int lane = threadIdx.x & 31;
  const __bf16* row =
      Kt + (size_t)(n0 + (lane & 15)) * ldk + k0 + ((lane >> 4) * 16);
  return *(const v16bf*)row;
}

// Fragment-linear swizzled B storage: element B[k][n] lives at
//   ((kt*ntiles + nt)*32 + lane)*16 + e,
// kt=k/32, nt=n/16, lane=(k%32/16)*16 + n%16, e=k%16.
// Loading a fragment = one contiguous 32B read per lane.
static __device__ __forceinline__ size_t bsw_index(int k, int n, int ntiles) {
  const int kt = k >> 5, kk = k & 31;
  const int lane = ((kk >> 4) << 4) + (n & 15);
  return ((size_t)((kt * ntiles + (n >> 4)) * 32 + lane) << 4) + (kk & 15);
}
static __device__ __forceinline__ v16bf load_bsw(const __bf16* Bsw, int ntiles,
                                                 int kt, int nt) {
  const int lane = threadIdx.x & 31;
  return *(const v16bf*)(Bsw +
                         (((size_t)(kt * ntiles + nt) * 32 + lane) << 4));
}

// Half-wave (16-lane) reductions: C-fragment rows live per 16-lane half.
static __device__ __forceinline__ float half16_max(float v) {
  v = fmaxf(v, __shfl_xor(v, 1));
  v = fmaxf(v, __shfl_xor(v, 2));
  v = fmaxf(v, __shfl_xor(v, 4));
  v = fmaxf(v, __shfl_xor(v, 8));
  return v;
}
static __device__ __forceinline__ float half16_sum(float v) {
  v += __shfl_xor(v, 1);
  v += __shfl_xor(v, 2);
  v += __shfl_xor(v, 4);
  v += __shfl_xor(v, 8);
  return v;
}

#define NEG_BIG (-1.0e30f)

// ------------------------------ prep kernels -------------------------------

__global__ void prep_x_kernel(const float* __restrict__ x,
                              __bf16* __restrict__ Xh, int n) {
  int i = blockIdx.x * blockDim.x + threadIdx.x;
  if (i < n) Xh[i] = (__bf16)x[i];
}

// Swizzled W for Q/K proj: logical B[k=d][n=p] = w[p][d] (p<PW else 0).
__global__ void prep_wqk_kernel(const float* __restrict__ w,
                                __bf16* __restrict__ W) {
  int i = blockIdx.x * blockDim.x + threadIdx.x;
  if (i >= DI * PP) return;
  int k = i / PP, n = i % PP;
  float val = (n < PW) ? w[(size_t)n * DI + k] : 0.0f;
  W[bsw_index(k, n, PP / 16)] = (__bf16)val;
}

// Swizzled Wv: logical B[k=d][n=e] = w_v[e][d].
__global__ void prep_wv_kernel(const float* __restrict__ w,
                               __bf16* __restrict__ W) {
  int i = blockIdx.x * blockDim.x + threadIdx.x;
  if (i >= DI * DI) return;
  int k = i / DI, n = i % DI;
  W[bsw_index(k, n, DI / 16)] = (__bf16)w[(size_t)n * DI + k];
}

// Kh (row-major): sink row 0 and zero tail rows (SQ+1 .. KPAD-1) per batch.
__global__ void prep_pad_k_kernel(const float* __restrict__ k0v,
                                  __bf16* __restrict__ Kh) {
  const int rowsPad = KPAD - SQ;  // 32
  int i = blockIdx.x * blockDim.x + threadIdx.x;
  if (i >= NB * rowsPad * PP) return;
  int p = i % PP;
  int t = i / PP;
  int row = t % rowsPad, b = t / rowsPad;
  int r = (row == 0) ? 0 : (SQ + row);
  float v = (row == 0 && p < PW) ? k0v[p] : 0.0f;
  Kh[((size_t)b * KPAD + r) * PP + p] = (__bf16)v;
}

// Vsw (swizzled, per batch: KPAD x DI): sink row + zero tail rows.
__global__ void prep_pad_v_kernel(const float* __restrict__ v0v,
                                  __bf16* __restrict__ Vsw) {
  const int rowsPad = KPAD - SQ;
  int i = blockIdx.x * blockDim.x + threadIdx.x;
  if (i >= NB * rowsPad * DI) return;
  int d = i % DI;
  int t = i / DI;
  int row = t % rowsPad, b = t / rowsPad;
  int r = (row == 0) ? 0 : (SQ + row);
  float v = (row == 0) ? v0v[d] : 0.0f;
  Vsw[(size_t)b * KPAD * DI + bsw_index(r, d, DI / 16)] = (__bf16)v;
}

// --------------------------- projection GEMMs ------------------------------

// Q/K: one wave computes a full 16 x 160 stripe (10 WMMAs per A fragment).
__global__ void qk_proj_kernel(const __bf16* __restrict__ Xh,
                               const __bf16* __restrict__ Wq,
                               const __bf16* __restrict__ Wk,
                               const float* __restrict__ bq,
                               const float* __restrict__ bk,
                               __bf16* __restrict__ Qh,
                               __bf16* __restrict__ Kh) {
  const int m0 = blockIdx.x * 16;        // global row over B*S
  const bool isK = blockIdx.y != 0;
  const __bf16* W = isK ? Wk : Wq;
  v8f acc[10] = {};
  for (int kt = 0; kt < DI / 32; ++kt) {
    v16bf a = load_a16(Xh, DI, m0, kt * 32);
#pragma unroll
    for (int t = 0; t < 10; ++t)
      acc[t] = wmma_bf16(a, load_bsw(W, 10, kt, t), acc[t]);
  }
  const int lane = threadIdx.x & 31;
  const int rbase = (lane >> 4) * 8;
#pragma unroll
  for (int t = 0; t < 10; ++t) {
    const int n = t * 16 + (lane & 15);
    const float bias = (n < PW) ? (isK ? bk[n] : bq[n]) : 0.0f;
#pragma unroll
    for (int r = 0; r < 8; ++r) {
      const int m = m0 + rbase + r;
      const int b_ = m >> 12;
      const int s = m & (SQ - 1);
      const float val = (n < PW) ? (acc[t][r] + bias) : 0.0f;
      if (isK)
        Kh[((size_t)b_ * KPAD + 1 + s) * PP + n] = (__bf16)val;
      else
        Qh[(size_t)m * PP + n] = (__bf16)val;
    }
  }
}

// V: one wave computes 16 x 64 (4 N-tiles), stores into swizzled Vsw.
__global__ void v_proj_kernel(const __bf16* __restrict__ Xh,
                              const __bf16* __restrict__ Wv,
                              const float* __restrict__ bv,
                              __bf16* __restrict__ Vsw) {
  const int m0 = blockIdx.x * 16;
  const int nt0 = blockIdx.y * 4;
  v8f acc[4] = {};
  for (int kt = 0; kt < DI / 32; ++kt) {
    v16bf a = load_a16(Xh, DI, m0, kt * 32);
#pragma unroll
    for (int t = 0; t < 4; ++t)
      acc[t] = wmma_bf16(a, load_bsw(Wv, DI / 16, kt, nt0 + t), acc[t]);
  }
  const int lane = threadIdx.x & 31;
  const int rbase = (lane >> 4) * 8;
#pragma unroll
  for (int t = 0; t < 4; ++t) {
    const int n = (nt0 + t) * 16 + (lane & 15);
    const float bias = bv[n];
#pragma unroll
    for (int r = 0; r < 8; ++r) {
      const int m = m0 + rbase + r;
      const int b_ = m >> 12;
      const int s = m & (SQ - 1);
      Vsw[(size_t)b_ * KPAD * DI + bsw_index(1 + s, n, DI / 16)] =
          (__bf16)(acc[t][r] + bias);
    }
  }
}

// --------------------- attention pass A: softmax stats ---------------------

__global__ void attn_stats_kernel(const __bf16* __restrict__ Qh,
                                  const __bf16* __restrict__ Kh,
                                  float* __restrict__ Mbuf,
                                  float* __restrict__ Lbuf) {
  const int qt = blockIdx.x;             // 0..1023
  const int b = qt >> 8;                 // 256 q-tiles per batch
  const int m0 = (qt & 255) * 16;
  const __bf16* Q = Qh + (size_t)b * SQ * PP;
  const __bf16* K = Kh + (size_t)b * KPAD * PP;
  const int lane = threadIdx.x & 31;
  const int rbase = (lane >> 4) * 8;

  v16bf qa[5];
#pragma unroll
  for (int c = 0; c < 5; ++c) qa[c] = load_a16(Q, PP, m0, c * 32);

  float rm[8], rl[8];
#pragma unroll
  for (int r = 0; r < 8; ++r) { rm[r] = NEG_BIG; rl[r] = 0.0f; }

  for (int j0 = 0; j0 < KPAD; j0 += 32) {
    if (j0 + 64 < KPAD)
      __builtin_prefetch(K + (size_t)(j0 + 64) * PP, 0, 1);
    v8f s0 = {}, s1 = {};
#pragma unroll
    for (int c = 0; c < 5; ++c) {
      v16bf b0 = load_bT16(K, PP, j0,      c * 32);
      v16bf b1 = load_bT16(K, PP, j0 + 16, c * 32);
      s0 = wmma_bf16(qa[c], b0, s0);
      s1 = wmma_bf16(qa[c], b1, s1);
    }
    const int col0 = j0 + (lane & 15);
    const int col1 = col0 + 16;
    const bool ok0 = col0 <= SQ;         // keys 0..4096 valid
    const bool ok1 = col1 <= SQ;
#pragma unroll
    for (int r = 0; r < 8; ++r) {
      const float v0 = ok0 ? s0[r] : NEG_BIG;
      const float v1 = ok1 ? s1[r] : NEG_BIG;
      const float mx = half16_max(fmaxf(v0, v1));
      const float mnew = fmaxf(rm[r], mx);
      const float scale = __expf(rm[r] - mnew);
      float p = __expf(v0 - mnew) + __expf(v1 - mnew);
      p = half16_sum(p);
      rl[r] = rl[r] * scale + p;
      rm[r] = mnew;
    }
  }
  if ((lane & 15) == 0) {
#pragma unroll
    for (int r = 0; r < 8; ++r) {
      const int s = m0 + rbase + r;
      Mbuf[(size_t)b * SQ + s] = rm[r];
      Lbuf[(size_t)b * SQ + s] = rl[r];
    }
  }
}

// ---------------- attention pass B: output per 128-col D chunk -------------

__global__ void attn_out_kernel(const float* __restrict__ x,
                                const __bf16* __restrict__ Qh,
                                const __bf16* __restrict__ Kh,
                                const __bf16* __restrict__ Vsw,
                                const float* __restrict__ Mbuf,
                                const float* __restrict__ Lbuf,
                                float* __restrict__ out) {
  __shared__ __align__(32) __bf16 Plds[16][40];  // 16x32 P tile, lda=40
  const int qt = blockIdx.x;
  const int b = qt >> 8;
  const int m0 = (qt & 255) * 16;
  const int d0 = blockIdx.y * 128;
  const __bf16* Q = Qh + (size_t)b * SQ * PP;
  const __bf16* K = Kh + (size_t)b * KPAD * PP;
  const __bf16* V = Vsw + (size_t)b * KPAD * DI;
  const int lane = threadIdx.x & 31;
  const int rbase = (lane >> 4) * 8;

  v16bf qa[5];
#pragma unroll
  for (int c = 0; c < 5; ++c) qa[c] = load_a16(Q, PP, m0, c * 32);

  float mrow[8], linv[8];
#pragma unroll
  for (int r = 0; r < 8; ++r) {
    const int s = m0 + rbase + r;
    mrow[r] = Mbuf[(size_t)b * SQ + s];
    linv[r] = 1.0f / Lbuf[(size_t)b * SQ + s];
  }

  v8f o[8] = {};
  for (int j0 = 0; j0 < KPAD; j0 += 32) {
    const int kt = j0 >> 5;
    if (j0 + 32 < KPAD)
      __builtin_prefetch(
          V + (((size_t)((kt + 1) * (DI / 16) + (d0 >> 4)) * 32) << 4), 0, 1);
    v8f s0 = {}, s1 = {};
#pragma unroll
    for (int c = 0; c < 5; ++c) {
      v16bf b0 = load_bT16(K, PP, j0,      c * 32);
      v16bf b1 = load_bT16(K, PP, j0 + 16, c * 32);
      s0 = wmma_bf16(qa[c], b0, s0);
      s1 = wmma_bf16(qa[c], b1, s1);
    }
    const int col0 = j0 + (lane & 15);
    const int col1 = col0 + 16;
#pragma unroll
    for (int r = 0; r < 8; ++r) {
      const float p0 = (col0 <= SQ) ? __expf(s0[r] - mrow[r]) : 0.0f;
      const float p1 = (col1 <= SQ) ? __expf(s1[r] - mrow[r]) : 0.0f;
      Plds[rbase + r][lane & 15]        = (__bf16)p0;
      Plds[rbase + r][(lane & 15) + 16] = (__bf16)p1;
    }
    __syncthreads();                     // single wave; DS in-order per wave
    v16bf pa = load_a16(&Plds[0][0], 40, 0, 0);
    __syncthreads();
#pragma unroll
    for (int nt = 0; nt < 8; ++nt) {
      v16bf vb = load_bsw(V, DI / 16, kt, (d0 >> 4) + nt);
      o[nt] = wmma_bf16(pa, vb, o[nt]);
    }
  }

#pragma unroll
  for (int nt = 0; nt < 8; ++nt) {
#pragma unroll
    for (int r = 0; r < 8; ++r) {
      const int s = m0 + rbase + r;
      const int d = d0 + nt * 16 + (lane & 15);
      const size_t idx = ((size_t)b * SQ + s) * DI + d;
      out[idx] = x[idx] + o[nt][r] * linv[r];
    }
  }
}

// ------------------------------- launcher ----------------------------------

extern "C" void kernel_launch(void* const* d_in, const int* in_sizes, int n_in,
                              void* d_out, int out_size, void* d_ws,
                              size_t ws_size, hipStream_t stream) {
  (void)in_sizes; (void)n_in; (void)out_size; (void)ws_size;
  const float* x   = (const float*)d_in[0];
  const float* w_q = (const float*)d_in[1];
  const float* b_q = (const float*)d_in[2];
  const float* w_k = (const float*)d_in[3];
  const float* b_k = (const float*)d_in[4];
  const float* w_v = (const float*)d_in[5];
  const float* b_v = (const float*)d_in[6];
  const float* k_0 = (const float*)d_in[7];
  const float* v_0 = (const float*)d_in[8];
  float* out = (float*)d_out;

  char* ws = (char*)d_ws;
  auto alloc = [&](size_t bytes) -> char* {
    char* p = ws;
    ws += (bytes + 255) & ~(size_t)255;
    return p;
  };
  __bf16* Xh  = (__bf16*)alloc((size_t)NB * SQ * DI * 2);    // x in bf16
  __bf16* Qh  = (__bf16*)alloc((size_t)NB * SQ * PP * 2);    // Q padded
  __bf16* Kh  = (__bf16*)alloc((size_t)NB * KPAD * PP * 2);  // K_full padded
  __bf16* Vsw = (__bf16*)alloc((size_t)NB * KPAD * DI * 2);  // V_full swizzled
  __bf16* Wq  = (__bf16*)alloc((size_t)DI * PP * 2);         // swizzled
  __bf16* Wk  = (__bf16*)alloc((size_t)DI * PP * 2);         // swizzled
  __bf16* Wv  = (__bf16*)alloc((size_t)DI * DI * 2);         // swizzled
  float*  Mb  = (float*)alloc((size_t)NB * SQ * 4);
  float*  Lb  = (float*)alloc((size_t)NB * SQ * 4);

  const int nx = NB * SQ * DI;
  prep_x_kernel<<<(nx + 255) / 256, 256, 0, stream>>>(x, Xh, nx);
  prep_wqk_kernel<<<(DI * PP + 255) / 256, 256, 0, stream>>>(w_q, Wq);
  prep_wqk_kernel<<<(DI * PP + 255) / 256, 256, 0, stream>>>(w_k, Wk);
  prep_wv_kernel<<<(DI * DI + 255) / 256, 256, 0, stream>>>(w_v, Wv);
  prep_pad_k_kernel<<<(NB * (KPAD - SQ) * PP + 255) / 256, 256, 0, stream>>>(
      k_0, Kh);
  prep_pad_v_kernel<<<(NB * (KPAD - SQ) * DI + 255) / 256, 256, 0, stream>>>(
      v_0, Vsw);

  qk_proj_kernel<<<dim3(NB * SQ / 16, 2), 32, 0, stream>>>(Xh, Wq, Wk, b_q,
                                                           b_k, Qh, Kh);
  v_proj_kernel<<<dim3(NB * SQ / 16, DI / 64), 32, 0, stream>>>(Xh, Wv, b_v,
                                                                Vsw);
  attn_stats_kernel<<<NB * SQ / 16, 32, 0, stream>>>(Qh, Kh, Mb, Lb);
  attn_out_kernel<<<dim3(NB * SQ / 16, DI / 128), 32, 0, stream>>>(
      x, Qh, Kh, Vsw, Mb, Lb, out);
}
